// Decoder_74904229642975
// MI455X (gfx1250) — matrix operand order repaired and checked by reading, not compiled, not run on hardware
//
#include <hip/hip_runtime.h>
#include <cstdint>
#include <cstddef>

// ---------------------------------------------------------------------------
// MI455X / gfx1250 implementation.
// Key algebraic simplification: self-attention sequence length is 1, so
// softmax over that axis == 1 exactly -> attn1 = V. Q/K of the self-attn
// QKV projection are never needed.
// Big GEMMs run through v_wmma_f32_16x16x32_f16 (f16 in, f32 accumulate).
// ---------------------------------------------------------------------------

typedef _Float16 f16;
typedef _Float16 v16h __attribute__((ext_vector_type(16)));
typedef _Float16 v8h  __attribute__((ext_vector_type(8)));
typedef float    v8f  __attribute__((ext_vector_type(8)));
typedef float    f32x4 __attribute__((ext_vector_type(4)));

#define BM 128
#define BN 128
#define BK 32
#define BKP 40   // padded LDS row stride (halfs): keeps b128 frag loads aligned

__device__ __forceinline__ float gelu_f(float v) {
    return 0.5f * v * (1.0f + erff(v * 0.70710678118654752f));
}

// A fragment: 16x32 f16. Lane L<16 holds row M=L, K={kb..kb+7, kb+16..kb+23},
// kb = 0 (lanes 0-15) or 8 (lanes 16-31) per CDNA5 ISA 7.12.2.
__device__ __forceinline__ v16h load_a_frag(const f16* lsA, int m0, int lane) {
    int m  = m0 + (lane & 15);
    int kb = (lane < 16) ? 0 : 8;
    const f16* p = lsA + m * BKP + kb;
    v8h lo = *(const v8h*)p;
    v8h hi = *(const v8h*)(p + 16);
    return __builtin_shufflevector(lo, hi, 0,1,2,3,4,5,6,7,8,9,10,11,12,13,14,15);
}

// B fragment: 32x16 f16, B staged transposed in LDS (per-column contiguous K).
// Lane L<16: column N=L, K={0..7,16..23}; lanes 16-31: column N=L-16, K={8..15,24..31}.
__device__ __forceinline__ v16h load_b_frag(const f16* lsB, int n0, int lane) {
    int n  = n0 + (lane & 15);
    int kb = (lane < 16) ? 0 : 8;
    const f16* p = lsB + n * BKP + kb;
    v8h lo = *(const v8h*)p;
    v8h hi = *(const v8h*)(p + 16);
    return __builtin_shufflevector(lo, hi, 0,1,2,3,4,5,6,7,8,9,10,11,12,13,14,15);
}

// ---------------------------------------------------------------------------
// Generic tiled WMMA GEMM:  C(MxN) = A(MxK,f32) @ B(KxN,f32 view at colOff,ldb)
//                            + bias, optional GELU, f32 or f16 output.
// Block 256 thr = 8 waves; each wave owns a 32x64 sub-tile (2x4 WMMA tiles).
// N and K must be multiples of 128 / 32 (true for every call below); M guarded.
// ---------------------------------------------------------------------------
__global__ __launch_bounds__(256) void gemm_wmma_kernel(
    const float* __restrict__ A, int lda,
    const float* __restrict__ B, int ldb, int colOff,
    const float* __restrict__ bias,
    float* __restrict__ Cf, f16* __restrict__ Ch,
    int M, int N, int K, int act /*0=none,1=gelu*/)
{
    __shared__ f16 lsA[BM * BKP];
    __shared__ f16 lsB[BN * BKP];

    const int t    = threadIdx.x;
    const int lane = t & 31;
    const int wave = t >> 5;
    const int wm   = wave & 3;          // wave row 0..3  (32 rows each)
    const int wn   = wave >> 2;         // wave col 0..1  (64 cols each)
    const int row0 = blockIdx.y * BM;
    const int col0 = blockIdx.x * BN;

    v8f acc[2][4];
#pragma unroll
    for (int i = 0; i < 2; i++)
#pragma unroll
        for (int j = 0; j < 4; j++)
#pragma unroll
            for (int e = 0; e < 8; e++) acc[i][j][e] = 0.0f;

    const int ac4 = (t & 7) * 4;   // A: col within [0,32)
    const int ar0 = t >> 3;        // A: row within [0,32), 4 passes
    const int bn0 = (t & 31) * 4;  // B: col within [0,128)
    const int bk0 = t >> 5;        // B: k within [0,8), 4 passes

    for (int k0 = 0; k0 < K; k0 += BK) {
#pragma unroll
        for (int i = 0; i < 4; i++) {
            int r = ar0 + i * 32;
            int grow = row0 + r;
            f32x4 v = {0.f, 0.f, 0.f, 0.f};
            if (grow < M) v = *(const f32x4*)(A + (size_t)grow * lda + k0 + ac4);
            f16* d = lsA + r * BKP + ac4;
            d[0] = (f16)v[0]; d[1] = (f16)v[1]; d[2] = (f16)v[2]; d[3] = (f16)v[3];
        }
#pragma unroll
        for (int i = 0; i < 4; i++) {
            int k = bk0 + i * 8;
            f32x4 v = *(const f32x4*)(B + (size_t)(k0 + k) * ldb + colOff + col0 + bn0);
            lsB[(bn0 + 0) * BKP + k] = (f16)v[0];
            lsB[(bn0 + 1) * BKP + k] = (f16)v[1];
            lsB[(bn0 + 2) * BKP + k] = (f16)v[2];
            lsB[(bn0 + 3) * BKP + k] = (f16)v[3];
        }
        __syncthreads();

        v16h af[2], bf[4];
#pragma unroll
        for (int i = 0; i < 2; i++) af[i] = load_a_frag(lsA, wm * 32 + i * 16, lane);
#pragma unroll
        for (int j = 0; j < 4; j++) bf[j] = load_b_frag(lsB, wn * 64 + j * 16, lane);
#pragma unroll
        for (int i = 0; i < 2; i++)
#pragma unroll
            for (int j = 0; j < 4; j++)
                acc[i][j] = __builtin_amdgcn_wmma_f32_16x16x32_f16(
                    false, af[i], false, bf[j], (short)0, acc[i][j], false, false);
        __syncthreads();
    }

    const int cn   = lane & 15;
    const int mofs = (lane < 16) ? 0 : 8;
#pragma unroll
    for (int i = 0; i < 2; i++) {
#pragma unroll
        for (int j = 0; j < 4; j++) {
            int gcol = col0 + wn * 64 + j * 16 + cn;
            float bv = bias ? bias[gcol] : 0.0f;
#pragma unroll
            for (int r = 0; r < 8; r++) {
                int grow = row0 + wm * 32 + i * 16 + mofs + r;
                if (grow < M) {
                    float v = acc[i][j][r] + bv;
                    if (act == 1) v = gelu_f(v);
                    if (Ch) Ch[(size_t)grow * N + gcol] = (f16)v;
                    else    Cf[(size_t)grow * N + gcol] = v;
                }
            }
        }
    }
}

// ---------------------------------------------------------------------------
// conv2 (3x3, 512->512, SAME) as implicit GEMM: K = 9*512, A rows gathered
// with per-tap spatial shift + zero pad. Epilogue fuses:
//   h = gelu(conv+b)*sqrt(512) + ex_enc[b] + spe[p]
// ---------------------------------------------------------------------------
__global__ __launch_bounds__(256) void conv2_wmma_kernel(
    const float* __restrict__ A,      // conv1 output, (14400 x 512)
    const float* __restrict__ W,      // (9*512 x 512)
    const float* __restrict__ bias,
    const float* __restrict__ exenc,  // (64 x 512)
    const float* __restrict__ spe,    // (225 x 512)
    float* __restrict__ Cf, int M)
{
    __shared__ f16 lsA[BM * BKP];
    __shared__ f16 lsB[BN * BKP];

    const int t    = threadIdx.x;
    const int lane = t & 31;
    const int wave = t >> 5;
    const int wm   = wave & 3;
    const int wn   = wave >> 2;
    const int row0 = blockIdx.y * BM;
    const int col0 = blockIdx.x * BN;
    const int N = 512, K = 9 * 512;

    v8f acc[2][4];
#pragma unroll
    for (int i = 0; i < 2; i++)
#pragma unroll
        for (int j = 0; j < 4; j++)
#pragma unroll
            for (int e = 0; e < 8; e++) acc[i][j][e] = 0.0f;

    const int ac4 = (t & 7) * 4;
    const int ar0 = t >> 3;
    const int bn0 = (t & 31) * 4;
    const int bk0 = t >> 5;

    for (int k0 = 0; k0 < K; k0 += BK) {
        int tap = k0 >> 9;            // k0 / 512 (BK divides 512 -> no straddle)
        int kin = k0 & 511;
        int dy = tap / 3 - 1, dx = tap % 3 - 1;
#pragma unroll
        for (int i = 0; i < 4; i++) {
            int r = ar0 + i * 32;
            int grow = row0 + r;
            f32x4 v = {0.f, 0.f, 0.f, 0.f};
            if (grow < M) {
                int b = grow / 225, p = grow % 225;
                int y = p / 15 + dy, x = p % 15 + dx;
                if (y >= 0 && y < 15 && x >= 0 && x < 15)
                    v = *(const f32x4*)(A + (size_t)(b * 225 + y * 15 + x) * 512 + kin + ac4);
            }
            f16* d = lsA + r * BKP + ac4;
            d[0] = (f16)v[0]; d[1] = (f16)v[1]; d[2] = (f16)v[2]; d[3] = (f16)v[3];
        }
#pragma unroll
        for (int i = 0; i < 4; i++) {
            int k = bk0 + i * 8;
            f32x4 v = *(const f32x4*)(W + (size_t)(k0 + k) * 512 + col0 + bn0);
            lsB[(bn0 + 0) * BKP + k] = (f16)v[0];
            lsB[(bn0 + 1) * BKP + k] = (f16)v[1];
            lsB[(bn0 + 2) * BKP + k] = (f16)v[2];
            lsB[(bn0 + 3) * BKP + k] = (f16)v[3];
        }
        __syncthreads();

        v16h af[2], bf[4];
#pragma unroll
        for (int i = 0; i < 2; i++) af[i] = load_a_frag(lsA, wm * 32 + i * 16, lane);
#pragma unroll
        for (int j = 0; j < 4; j++) bf[j] = load_b_frag(lsB, wn * 64 + j * 16, lane);
#pragma unroll
        for (int i = 0; i < 2; i++)
#pragma unroll
            for (int j = 0; j < 4; j++)
                acc[i][j] = __builtin_amdgcn_wmma_f32_16x16x32_f16(
                    false, af[i], false, bf[j], (short)0, acc[i][j], false, false);
        __syncthreads();
    }

    const int cn   = lane & 15;
    const int mofs = (lane < 16) ? 0 : 8;
    const float scale = 22.62741699796952f;   // sqrt(512)
#pragma unroll
    for (int i = 0; i < 2; i++) {
#pragma unroll
        for (int j = 0; j < 4; j++) {
            int gcol = col0 + wn * 64 + j * 16 + cn;
            float bv = bias[gcol];
#pragma unroll
            for (int r = 0; r < 8; r++) {
                int grow = row0 + wm * 32 + i * 16 + mofs + r;
                if (grow < M) {
                    int b = grow / 225, p = grow % 225;
                    float v = gelu_f(acc[i][j][r] + bv) * scale
                              + exenc[(size_t)b * 512 + gcol] + spe[(size_t)p * 512 + gcol];
                    Cf[(size_t)grow * 512 + gcol] = v;
                }
            }
        }
    }
}

// ---------------------------------------------------------------------------
// conv1: 3x3, 2->512 channels, GELU. One WG per output position.
// ---------------------------------------------------------------------------
__global__ __launch_bounds__(256) void conv1_kernel(
    const float* __restrict__ x, const float* __restrict__ w,
    const float* __restrict__ bias, float* __restrict__ out)
{
    __shared__ float sx[18];
    int r = blockIdx.x, t = threadIdx.x;
    int b = r / 225, p = r % 225, y = p / 15, xx = p % 15;
    if (t < 18) {
        int tap = t >> 1, ci = t & 1;
        int yy = y + tap / 3 - 1, xc = xx + tap % 3 - 1;
        float v = 0.f;
        if (yy >= 0 && yy < 15 && xc >= 0 && xc < 15)
            v = x[(size_t)(b * 225 + yy * 15 + xc) * 2 + ci];
        sx[t] = v;
    }
    __syncthreads();
    for (int co = t; co < 512; co += 256) {
        float a = bias[co];
#pragma unroll
        for (int q = 0; q < 18; q++) a += sx[q] * w[q * 512 + co];
        out[(size_t)r * 512 + co] = gelu_f(a);
    }
}

// ---------------------------------------------------------------------------
// ex MLP: tanh((ex@W1+b1)@W2+b2). One WG per batch row (64 total).
// ---------------------------------------------------------------------------
__global__ __launch_bounds__(256) void exmlp_kernel(
    const float* __restrict__ ex, const float* __restrict__ w1,
    const float* __restrict__ b1, const float* __restrict__ w2,
    const float* __restrict__ b2, float* __restrict__ out)
{
    __shared__ float se[64];
    __shared__ float sh[2048];
    int b = blockIdx.x, t = threadIdx.x;
    if (t < 64) se[t] = ex[b * 64 + t];
    __syncthreads();
    for (int j = t; j < 2048; j += 256) {
        float a = b1[j];
#pragma unroll 8
        for (int k = 0; k < 64; k++) a += se[k] * w1[k * 2048 + j];
        sh[j] = a;
    }
    __syncthreads();
    for (int d = t; d < 512; d += 256) {
        float a = b2[d];
        for (int j = 0; j < 2048; j++) a += sh[j] * w2[j * 512 + d];
        out[(size_t)b * 512 + d] = tanhf(a);
    }
}

// ---------------------------------------------------------------------------
// SPE positional encoding (225 x 512).
// ---------------------------------------------------------------------------
__global__ __launch_bounds__(256) void spe_kernel(float* __restrict__ spe)
{
    int p = blockIdx.x, t = threadIdx.x;
    float pr = (float)(p / 15 - 7);
    float pc = (float)(p % 15 - 7);
    for (int c = t; c < 512; c += 256) {
        int i2 = c & ~1;
        float rate = expf(-(float)i2 * (9.210340371976184f / 512.0f)); // 10000^{-2(c/2)/512}
        spe[(size_t)p * 512 + c] = (c & 1) ? cosf(pc * rate) : sinf(pr * rate);
    }
}

// ---------------------------------------------------------------------------
// Cross attention (S=12): logits = q_h . K_h / 8 + mask, softmax, out = p @ V.
// One WG per token row; K/V pre-materialized in f16 (R*12 x 1024).
// ---------------------------------------------------------------------------
__global__ __launch_bounds__(256) void attn_kernel(
    const float* __restrict__ q2, const f16* __restrict__ kv,
    const float* __restrict__ pmask, float* __restrict__ out)
{
    __shared__ float sq[512];
    __shared__ f16   skv[12 * 1024];
    __shared__ float sl[96];
    __shared__ float sp[96];
    int r = blockIdx.x, t = threadIdx.x;
    sq[t]       = q2[(size_t)r * 512 + t];
    sq[t + 256] = q2[(size_t)r * 512 + 256 + t];
    const uint32_t* src = (const uint32_t*)(kv + (size_t)r * 12288);
    uint32_t* dst = (uint32_t*)skv;
    for (int i = t; i < 6144; i += 256) dst[i] = src[i];
    __syncthreads();
    if (t < 96) {
        int hh = t / 12, s = t % 12;
        const f16* kk = skv + s * 1024 + hh * 64;
        const float* qq = sq + hh * 64;
        float a = 0.f;
#pragma unroll
        for (int d = 0; d < 64; d++) a += qq[d] * (float)kk[d];
        sl[t] = a * 0.125f + pmask[s] * (-1e9f);
    }
    __syncthreads();
    if (t < 8) {
        float mx = -1e30f;
        for (int s = 0; s < 12; s++) mx = fmaxf(mx, sl[t * 12 + s]);
        float e[12], sum = 0.f;
        for (int s = 0; s < 12; s++) { e[s] = expf(sl[t * 12 + s] - mx); sum += e[s]; }
        float inv = 1.0f / sum;
        for (int s = 0; s < 12; s++) sp[t * 12 + s] = e[s] * inv;
    }
    __syncthreads();
    for (int c = t; c < 512; c += 256) {
        int hh = c >> 6;
        float a = 0.f;
#pragma unroll
        for (int s = 0; s < 12; s++) a += sp[hh * 12 + s] * (float)skv[s * 1024 + 512 + c];
        out[(size_t)r * 512 + c] = a;
    }
}

// ---------------------------------------------------------------------------
// out = LayerNorm(x + res) * g + beta, per 512-wide row.
// ---------------------------------------------------------------------------
__global__ __launch_bounds__(256) void add_ln_kernel(
    const float* __restrict__ x, const float* __restrict__ res,
    const float* __restrict__ g, const float* __restrict__ beta,
    float* __restrict__ out)
{
    __shared__ float red[256];
    int r = blockIdx.x, t = threadIdx.x;
    size_t base = (size_t)r * 512;
    float v0 = x[base + t] + res[base + t];
    float v1 = x[base + t + 256] + res[base + t + 256];
    red[t] = v0 + v1;
    __syncthreads();
    for (int s = 128; s > 0; s >>= 1) { if (t < s) red[t] += red[t + s]; __syncthreads(); }
    float mean = red[0] * (1.0f / 512.0f);
    __syncthreads();
    float d0 = v0 - mean, d1 = v1 - mean;
    red[t] = d0 * d0 + d1 * d1;
    __syncthreads();
    for (int s = 128; s > 0; s >>= 1) { if (t < s) red[t] += red[t + s]; __syncthreads(); }
    float rstd = rsqrtf(red[0] * (1.0f / 512.0f) + 1e-6f);
    out[base + t]       = d0 * rstd * g[t] + beta[t];
    out[base + t + 256] = d1 * rstd * g[t + 256] + beta[t + 256];
}

// ---------------------------------------------------------------------------
// Host-side orchestration.
// ---------------------------------------------------------------------------
extern "C" void kernel_launch(void* const* d_in, const int* in_sizes, int n_in,
                              void* d_out, int out_size, void* d_ws, size_t ws_size,
                              hipStream_t stream)
{
    (void)in_sizes; (void)n_in; (void)out_size; (void)ws_size;

    const float* x     = (const float*)d_in[0];
    const float* ex    = (const float*)d_in[1];
    const float* enc   = (const float*)d_in[2];
    const float* pmask = (const float*)d_in[4];      // padding mask (12)
    const float* ex_w1 = (const float*)d_in[5];
    const float* ex_b1 = (const float*)d_in[6];
    const float* ex_w2 = (const float*)d_in[7];
    const float* ex_b2 = (const float*)d_in[8];
    const float* cw1   = (const float*)d_in[9];
    const float* cb1   = (const float*)d_in[10];
    const float* cw2   = (const float*)d_in[11];
    const float* cb2   = (const float*)d_in[12];
    const float* wx_w  = (const float*)d_in[13];
    const float* wx_b  = (const float*)d_in[14];
    const float* wo1_w = (const float*)d_in[15];
    const float* wo1_b = (const float*)d_in[16];
    const float* wq_w  = (const float*)d_in[17];
    const float* wq_b  = (const float*)d_in[18];
    const float* wkv_w = (const float*)d_in[19];
    const float* wkv_b = (const float*)d_in[20];
    const float* wo2_w = (const float*)d_in[21];
    const float* wo2_b = (const float*)d_in[22];
    const float* fw1   = (const float*)d_in[23];
    const float* fb1   = (const float*)d_in[24];
    const float* fw2   = (const float*)d_in[25];
    const float* fb2   = (const float*)d_in[26];
    const float* ln_g  = (const float*)d_in[27];
    const float* ln_b  = (const float*)d_in[28];

    const int R  = 64 * 225;     // 14400 token rows
    const int R2 = R * 12;       // 172800 encoder rows

    // workspace layout (floats, then halfs)
    float* ws    = (float*)d_ws;
    float* h     = ws;                            // R*512
    float* t1    = h    + (size_t)R * 512;        // R*512 (conv1 out / out1)
    float* t2    = t1   + (size_t)R * 512;        // R*512
    float* t3    = t2   + (size_t)R * 512;        // R*512
    float* ffnb  = t3   + (size_t)R * 512;        // R*2048
    float* spe   = ffnb + (size_t)R * 2048;       // 225*512
    float* exenc = spe  + (size_t)225 * 512;      // 64*512
    f16*   kv16  = (f16*)(exenc + (size_t)64 * 512);  // R2*1024 halfs

    spe_kernel<<<225, 256, 0, stream>>>(spe);
    exmlp_kernel<<<64, 256, 0, stream>>>(ex, ex_w1, ex_b1, ex_w2, ex_b2, exenc);
    conv1_kernel<<<R, 256, 0, stream>>>(x, cw1, cb1, t1);

    dim3 gC(4, (R + BM - 1) / BM);
    conv2_wmma_kernel<<<gC, 256, 0, stream>>>(t1, cw2, cb2, exenc, spe, h, R);

    dim3 g512(4,  (R  + BM - 1) / BM);
    dim3 gKV (8,  (R2 + BM - 1) / BM);
    dim3 g2048(16,(R  + BM - 1) / BM);

    for (int i = 0; i < 4; i++) {
        const float* Wx  = wx_w  + (size_t)i * 512 * 1536;  // V = cols [1024,1536)
        const float* bv  = wx_b  + (size_t)i * 1536 + 1024;
        const float* Wo1 = wo1_w + (size_t)i * 512 * 512;
        const float* bo1 = wo1_b + (size_t)i * 512;
        const float* Wq  = wq_w  + (size_t)i * 512 * 512;
        const float* bq  = wq_b  + (size_t)i * 512;
        const float* Wkv = wkv_w + (size_t)i * 512 * 1024;
        const float* bkv = wkv_b + (size_t)i * 1024;
        const float* Wo2 = wo2_w + (size_t)i * 512 * 512;
        const float* bo2 = wo2_b + (size_t)i * 512;
        const float* W1  = fw1   + (size_t)i * 512 * 2048;
        const float* B1  = fb1   + (size_t)i * 2048;
        const float* W2  = fw2   + (size_t)i * 2048 * 512;
        const float* B2  = fb2   + (size_t)i * 512;
        const float* g0  = ln_g + (size_t)(i * 3 + 0) * 512;
        const float* b0  = ln_b + (size_t)(i * 3 + 0) * 512;
        const float* g1  = ln_g + (size_t)(i * 3 + 1) * 512;
        const float* b1l = ln_b + (size_t)(i * 3 + 1) * 512;
        const float* g2  = ln_g + (size_t)(i * 3 + 2) * 512;
        const float* b2l = ln_b + (size_t)(i * 3 + 2) * 512;

        // self-attn (T=1 => softmax==1 => attn == V):  v = h @ Wv + bv
        gemm_wmma_kernel<<<g512, 256, 0, stream>>>(h, 512, Wx, 1536, 1024, bv,
                                                   t2, (f16*)nullptr, R, 512, 512, 0);
        // attn1 = v @ Wo1 + bo1
        gemm_wmma_kernel<<<g512, 256, 0, stream>>>(t2, 512, Wo1, 512, 0, bo1,
                                                   t3, (f16*)nullptr, R, 512, 512, 0);
        // out1 = LN(attn1 + h)
        add_ln_kernel<<<R, 256, 0, stream>>>(t3, h, g0, b0, t1);
        // q2 = out1 @ Wq + bq
        gemm_wmma_kernel<<<g512, 256, 0, stream>>>(t1, 512, Wq, 512, 0, bq,
                                                   t2, (f16*)nullptr, R, 512, 512, 0);
        // KV = enc @ Wkv + bkv  (dominant GEMM; f16 output halves HBM traffic)
        gemm_wmma_kernel<<<gKV, 256, 0, stream>>>(enc, 512, Wkv, 1024, 0, bkv,
                                                  (float*)nullptr, kv16, R2, 1024, 512, 0);
        // o = softmax(q2 K^T / 8 + mask) V
        attn_kernel<<<R, 256, 0, stream>>>(t2, kv16, pmask, t3);
        // attn2 = o @ Wo2 + bo2
        gemm_wmma_kernel<<<g512, 256, 0, stream>>>(t3, 512, Wo2, 512, 0, bo2,
                                                   t2, (f16*)nullptr, R, 512, 512, 0);
        // out2 = LN(attn2 + out1)
        add_ln_kernel<<<R, 256, 0, stream>>>(t2, t1, g1, b1l, t3);
        // ffn hidden = gelu(out2 @ W1 + B1)
        gemm_wmma_kernel<<<g2048, 256, 0, stream>>>(t3, 512, W1, 2048, 0, B1,
                                                    ffnb, (f16*)nullptr, R, 2048, 512, 1);
        // ffn out = hidden @ W2 + B2
        gemm_wmma_kernel<<<g512, 256, 0, stream>>>(ffnb, 2048, W2, 512, 0, B2,
                                                   t2, (f16*)nullptr, R, 512, 2048, 0);
        // h = LN(ffn + out2)   (last layer writes d_out directly)
        float* dst = (i == 3) ? (float*)d_out : h;
        add_ln_kernel<<<R, 256, 0, stream>>>(t2, t3, g2, b2l, dst);
    }
}
